// NPTLayer_27805618274544
// MI455X (gfx1250) — matrix operand order repaired and checked by reading, not compiled, not run on hardware
//
#include <hip/hip_runtime.h>

#define DMODEL 2048
#define FDIM   8192
#define RLORA  16
#define NHEAD  16
#define HDIM   128
#define SEQ    2048
#define BATCH  2

#ifndef __has_builtin
#define __has_builtin(x) 0
#endif

#if __has_builtin(__builtin_amdgcn_global_load_async_to_lds_b128)
#define HAVE_ASYNC 1
#else
#define HAVE_ASYNC 0
#endif

#if HAVE_ASYNC
#if __has_builtin(__builtin_amdgcn_s_wait_asynccnt)
#define WAIT_ASYNC() __builtin_amdgcn_s_wait_asynccnt(0)
#else
#define WAIT_ASYNC() asm volatile("s_wait_asynccnt 0x0" ::: "memory")
#endif
typedef int gv4i __attribute__((vector_size(16)));
typedef __attribute__((address_space(1))) gv4i* gptr_v4i;
typedef __attribute__((address_space(3))) gv4i* lptr_v4i;
// global (as1) -> LDS (as3) 16-byte async DMA copy, per-lane addresses
__device__ __forceinline__ void async_b128(const void* g, void* l) {
    __builtin_amdgcn_global_load_async_to_lds_b128(
        (gptr_v4i)(unsigned long long)(uintptr_t)g,
        (lptr_v4i)(unsigned)(uintptr_t)l,
        0, 0);
}
#endif

typedef __attribute__((ext_vector_type(16))) __bf16 v16bf;
typedef __attribute__((ext_vector_type(8)))  float  v8f;

union FragB {
    v16bf v;
    uint4 q[2];
    unsigned short u[16];
};

__device__ __forceinline__ unsigned short f2bf(float f) {
    union { float f; unsigned u; } c; c.f = f;
    unsigned u = c.u;
    u += 0x7fffu + ((u >> 16) & 1u);            // round-to-nearest-even
    return (unsigned short)(u >> 16);
}
__device__ __forceinline__ float bf2f(unsigned short h) {
    union { unsigned u; float f; } c; c.u = ((unsigned)h) << 16;
    return c.f;
}

// ---------------------------------------------------------------------------
// RMSNorm (fp32 in) -> bf16 out.  One block per token row (D=2048, 256 thr).
// ---------------------------------------------------------------------------
__global__ __launch_bounds__(256) void rmsnorm_bf16(const float* __restrict__ x,
                                                    const float* __restrict__ w,
                                                    unsigned short* __restrict__ out) {
    const int row = blockIdx.x;
    const float* xr = x + (size_t)row * DMODEL;
    unsigned short* orow = out + (size_t)row * DMODEL;
    const int t = threadIdx.x;
    float vals[8];
    float ss = 0.f;
#pragma unroll
    for (int i = 0; i < 8; ++i) {
        vals[i] = xr[t + i * 256];
        ss += vals[i] * vals[i];
    }
#pragma unroll
    for (int off = 16; off > 0; off >>= 1) ss += __shfl_down(ss, off, 32);
    __shared__ float red[8];
    if ((t & 31) == 0) red[t >> 5] = ss;
    __syncthreads();
    float tot = 0.f;
#pragma unroll
    for (int i = 0; i < 8; ++i) tot += red[i];
    const float scale = rsqrtf(tot / (float)DMODEL + 1e-6f);
#pragma unroll
    for (int i = 0; i < 8; ++i)
        orow[t + i * 256] = f2bf(vals[i] * scale * w[t + i * 256]);
}

// ---------------------------------------------------------------------------
// bf16 WMMA GEMM:  C[M,N] = A[M,K](bf16) @ W[N,K](fp32->bf16)^T
// Block tile 128x128, 8 waves, wave tile 64x32 (4x2 WMMA 16x16 tiles), K=32 step.
// A tile: async global->LDS DMA with ping-pong buffers (ASYNCcnt) when available,
// else register double-buffer.  W tile: register double-buffer + f32->bf16 convert.
// Epilogues: 0 = store bf16, 2 = LoRA + SiLU * up (bf16), 3 = + residual (f32).
// ---------------------------------------------------------------------------
enum { EPI_BF16 = 0, EPI_GATE = 2, EPI_RESID = 3 };

template <int EPI>
__global__ __launch_bounds__(256) void gemm_bf16(
        const unsigned short* __restrict__ Abuf,   // M x K bf16
        const float*          __restrict__ Wbuf,   // N x K f32 (torch Linear weight)
        void*                 __restrict__ Cbuf,
        int M, int N, int K,
        const float*          __restrict__ modhA,  // M x 16   (GATE)
        const float*          __restrict__ Blora,  // 16 x N   (GATE)
        const unsigned short* __restrict__ upbuf,  // M x N bf16 (GATE)
        const float*          __restrict__ resid)  // M x N f32  (RESID)
{
    __shared__ unsigned short As[2][128][48];
    __shared__ unsigned short Bs[128][48];
    __shared__ float mh[128][16];                   // only used by EPI_GATE

    const int nBase = blockIdx.x * 128;
    const int mBase = blockIdx.y * 128;
    const int t    = threadIdx.x;
    const int wave = t >> 5;
    const int lane = t & 31;
    const int wm = (wave & 1) * 64;
    const int wn = (wave >> 1) * 32;
    const int r  = lane & 15;
    const int hh = lane >> 4;

    const v8f vz = {0.f, 0.f, 0.f, 0.f, 0.f, 0.f, 0.f, 0.f};
    v8f acc[4][2];
#pragma unroll
    for (int i = 0; i < 4; ++i) { acc[i][0] = vz; acc[i][1] = vz; }

    const int ldr = t >> 1;
    const int lk0 = (t & 1) * 16;
    const unsigned short* agp = Abuf + (size_t)(mBase + ldr) * K + lk0;
    const float*          wgp = Wbuf + (size_t)(nBase + ldr) * K + lk0;

    float4 rw0, rw1, rw2, rw3;
    auto load_globalW = [&](int kb) {
        const float4* ws = (const float4*)(wgp + kb);
        rw0 = ws[0]; rw1 = ws[1]; rw2 = ws[2]; rw3 = ws[3];
        if (kb + 32 < K) {
            __builtin_prefetch(agp + kb + 32, 0, 1);
            __builtin_prefetch(wgp + kb + 32, 0, 1);
        }
    };
    auto store_ldsW = [&]() {
        unsigned int* bd = (unsigned int*)(&Bs[ldr][lk0]);
        bd[0] = (unsigned)f2bf(rw0.x) | ((unsigned)f2bf(rw0.y) << 16);
        bd[1] = (unsigned)f2bf(rw0.z) | ((unsigned)f2bf(rw0.w) << 16);
        bd[2] = (unsigned)f2bf(rw1.x) | ((unsigned)f2bf(rw1.y) << 16);
        bd[3] = (unsigned)f2bf(rw1.z) | ((unsigned)f2bf(rw1.w) << 16);
        bd[4] = (unsigned)f2bf(rw2.x) | ((unsigned)f2bf(rw2.y) << 16);
        bd[5] = (unsigned)f2bf(rw2.z) | ((unsigned)f2bf(rw2.w) << 16);
        bd[6] = (unsigned)f2bf(rw3.x) | ((unsigned)f2bf(rw3.y) << 16);
        bd[7] = (unsigned)f2bf(rw3.z) | ((unsigned)f2bf(rw3.w) << 16);
    };
    auto compute = [&](int buf) {
        FragB af[4], bfg[2];
#pragma unroll
        for (int mt = 0; mt < 4; ++mt) {
            // 16-bit A 16x32 layout: q[0] = K [hh*8 .. +7], q[1] = K [16+hh*8 .. +7]
            const uint4* p0 = (const uint4*)(&As[buf][wm + mt * 16 + r][hh * 8]);
            const uint4* p1 = (const uint4*)(&As[buf][wm + mt * 16 + r][16 + hh * 8]);
            af[mt].q[0] = p0[0];
            af[mt].q[1] = p1[0];
        }
#pragma unroll
        for (int nt = 0; nt < 2; ++nt) {
            // B 32x16 layout: lane(n)=r, q[0] = K [hh*8 .. +7], q[1] = K [16+hh*8 .. +7]
            const uint4* p0 = (const uint4*)(&Bs[wn + nt * 16 + r][hh * 8]);
            const uint4* p1 = (const uint4*)(&Bs[wn + nt * 16 + r][16 + hh * 8]);
            bfg[nt].q[0] = p0[0];
            bfg[nt].q[1] = p1[0];
        }
#pragma unroll
        for (int mt = 0; mt < 4; ++mt)
#pragma unroll
            for (int nt = 0; nt < 2; ++nt)
                acc[mt][nt] = __builtin_amdgcn_wmma_f32_16x16x32_bf16(
                    false, af[mt].v, false, bfg[nt].v, (short)0, acc[mt][nt],
                    false, false);
    };

#if HAVE_ASYNC
    auto async_loadA = [&](int kb, int buf) {
        const unsigned short* g = agp + kb;
        unsigned short* l = &As[buf][ldr][lk0];
        async_b128(g, l);
        async_b128(g + 8, l + 8);
    };
    async_loadA(0, 0);
    load_globalW(0);
    store_ldsW();
    WAIT_ASYNC();
    __syncthreads();
    for (int kb = 0; kb < K; kb += 32) {
        const int  buf  = (kb >> 5) & 1;
        const bool more = (kb + 32) < K;
        if (more) {
            async_loadA(kb + 32, buf ^ 1);   // DMA next A tile while we compute
            load_globalW(kb + 32);
        }
        compute(buf);
        __syncthreads();
        if (more) {
            store_ldsW();
            WAIT_ASYNC();
            __syncthreads();
        }
    }
#else
    uint4 ra0, ra1;
    auto load_globalA = [&](int kb) {
        const uint4* src = (const uint4*)(agp + kb);
        ra0 = src[0];
        ra1 = src[1];
    };
    auto store_ldsA = [&]() {
        uint4* dst = (uint4*)(&As[0][ldr][lk0]);
        dst[0] = ra0;
        dst[1] = ra1;
    };
    load_globalA(0);
    load_globalW(0);
    store_ldsA();
    store_ldsW();
    __syncthreads();
    for (int kb = 0; kb < K; kb += 32) {
        const bool more = (kb + 32) < K;
        if (more) { load_globalA(kb + 32); load_globalW(kb + 32); }
        compute(0);
        __syncthreads();
        if (more) {
            store_ldsA();
            store_ldsW();
            __syncthreads();
        }
    }
#endif

    // ---------------- epilogues ----------------
    if (EPI == EPI_GATE) {
#pragma unroll
        for (int i = 0; i < 8; ++i) {
            int idx = t * 8 + i;                 // 128*16 = 2048 floats
            mh[idx >> 4][idx & 15] =
                modhA[(size_t)(mBase + (idx >> 4)) * RLORA + (idx & 15)];
        }
        __syncthreads();
        unsigned short* C = (unsigned short*)Cbuf;
#pragma unroll
        for (int nt = 0; nt < 2; ++nt) {
            const int n = nBase + wn + nt * 16 + r;
            float bc[16];
#pragma unroll
            for (int rr = 0; rr < 16; ++rr) bc[rr] = Blora[(size_t)rr * N + n];
#pragma unroll
            for (int mt = 0; mt < 4; ++mt) {
#pragma unroll
                for (int i = 0; i < 8; ++i) {
                    const int rowl = wm + mt * 16 + i + hh * 8;
                    float x = acc[mt][nt][i];
#pragma unroll
                    for (int rr = 0; rr < 16; ++rr) x += mh[rowl][rr] * bc[rr];
                    const float g = x / (1.f + __expf(-x));            // SiLU
                    const float u = bf2f(upbuf[(size_t)(mBase + rowl) * N + n]);
                    C[(size_t)(mBase + rowl) * N + n] = f2bf(g * u);
                }
            }
        }
    } else if (EPI == EPI_RESID) {
        float* C = (float*)Cbuf;
#pragma unroll
        for (int nt = 0; nt < 2; ++nt) {
            const int n = nBase + wn + nt * 16 + r;
#pragma unroll
            for (int mt = 0; mt < 4; ++mt)
#pragma unroll
                for (int i = 0; i < 8; ++i) {
                    const int rowl = wm + mt * 16 + i + hh * 8;
                    const size_t idx = (size_t)(mBase + rowl) * N + n;
                    C[idx] = acc[mt][nt][i] + resid[idx];
                }
        }
    } else {
        unsigned short* C = (unsigned short*)Cbuf;
#pragma unroll
        for (int nt = 0; nt < 2; ++nt) {
            const int n = nBase + wn + nt * 16 + r;
#pragma unroll
            for (int mt = 0; mt < 4; ++mt)
#pragma unroll
                for (int i = 0; i < 8; ++i) {
                    const int rowl = wm + mt * 16 + i + hh * 8;
                    C[(size_t)(mBase + rowl) * N + n] = f2bf(acc[mt][nt][i]);
                }
        }
    }
}

// ---------------------------------------------------------------------------
// Flash attention, bf16 WMMA.  Block = (q-tile 64, head, batch), 256 threads.
// Q/K tiles loaded with async global->LDS DMA when available.
// S=QK^T via WMMA (HD=128 -> 4 K-steps), online softmax (4 thr/row + shfl),
// V kept TRANSPOSED in LDS so P@V B-fragments are contiguous ds_load_b128,
// O accumulated in WMMA C fragments (rescaled per-row by alpha).
// ---------------------------------------------------------------------------
__global__ __launch_bounds__(256) void flash_attn(
        const unsigned short* __restrict__ qb,
        const unsigned short* __restrict__ kb,
        const unsigned short* __restrict__ vb,
        unsigned short* __restrict__ ob) {
    const int qt = blockIdx.x, h = blockIdx.y, b = blockIdx.z;
    const int qBase = qt * 64;

    extern __shared__ char smem[];
    unsigned short* Qs = (unsigned short*)smem;      // [64][136]
    unsigned short* Ks = Qs + 64 * 136;              // [64][136]
    unsigned short* Vt = Ks + 64 * 136;              // [128][72]  V TRANSPOSED
    unsigned short* Ps = Vt + 128 * 72;              // [64][72]   bf16 probs
    float* Ss   = (float*)(Ps + 64 * 72);            // [64][72]   scores
    float* mrow = Ss + 64 * 72;
    float* lrow = mrow + 64;
    float* arow = lrow + 64;

    const int t = threadIdx.x, wave = t >> 5, lane = t & 31;
    const int r = lane & 15, hh = lane >> 4;
    const size_t hoff  = (size_t)h * HDIM;
    const size_t bbase = (size_t)b * SEQ * DMODEL;
    const int lr = t >> 2, c0 = (t & 3) * 32;

    {   // load Q tile (64 x 128 bf16)
        const unsigned short* gq = qb + bbase + (size_t)(qBase + lr) * DMODEL + hoff + c0;
        unsigned short* lq = Qs + lr * 136 + c0;
#if HAVE_ASYNC
        async_b128(gq,      lq);
        async_b128(gq + 8,  lq + 8);
        async_b128(gq + 16, lq + 16);
        async_b128(gq + 24, lq + 24);
#else
        const uint4* src = (const uint4*)gq;
        uint4* dst = (uint4*)lq;
        dst[0] = src[0]; dst[1] = src[1]; dst[2] = src[2]; dst[3] = src[3];
#endif
    }
    if (t < 64) { mrow[t] = -3.0e38f; lrow[t] = 0.f; }

    const v8f vz = {0.f, 0.f, 0.f, 0.f, 0.f, 0.f, 0.f, 0.f};
    v8f oacc[4];
#pragma unroll
    for (int i = 0; i < 4; ++i) oacc[i] = vz;

    const int omt = wave >> 1;           // O / S tile row (0..3)
    const int onb = (wave & 1) * 64;     // O col base (0 or 64)
    const int sn0 = (wave & 1) * 2;      // S tile col pair
    const float rs = 0.08838834764831845f;  // 1/sqrt(128)

    for (int kt = 0; kt <= qt; ++kt) {
        const int kBase = kt * 64;
        __syncthreads();
        {   // K tile via async DMA; V tile via VGPRs + transpose into Vt[col][key]
            const unsigned short* gk = kb + bbase + (size_t)(kBase + lr) * DMODEL + hoff + c0;
            unsigned short* lk = Ks + lr * 136 + c0;
#if HAVE_ASYNC
            async_b128(gk,      lk);
            async_b128(gk + 8,  lk + 8);
            async_b128(gk + 16, lk + 16);
            async_b128(gk + 24, lk + 24);
#else
            const uint4* sk = (const uint4*)gk;
            uint4* dk = (uint4*)lk;
            dk[0] = sk[0]; dk[1] = sk[1]; dk[2] = sk[2]; dk[3] = sk[3];
#endif
            union { uint4 q[4]; unsigned short u[32]; } vl;
            const uint4* sv = (const uint4*)(vb + bbase + (size_t)(kBase + lr) * DMODEL + hoff + c0);
            vl.q[0] = sv[0]; vl.q[1] = sv[1]; vl.q[2] = sv[2]; vl.q[3] = sv[3];
#pragma unroll
            for (int j = 0; j < 32; ++j)
                Vt[(c0 + j) * 72 + lr] = vl.u[j];
#if HAVE_ASYNC
            WAIT_ASYNC();
#endif
        }
        __syncthreads();

        // ---- S = Q @ K^T * rs, causal mask, to LDS ----
#pragma unroll
        for (int sni = 0; sni < 2; ++sni) {
            const int snt = sn0 + sni;
            v8f sacc = vz;
#pragma unroll
            for (int kc = 0; kc < 4; ++kc) {
                FragB aq, bk;
                const uint4* pa0 = (const uint4*)(Qs + (omt * 16 + r) * 136 + kc * 32 + hh * 8);
                const uint4* pa1 = (const uint4*)(Qs + (omt * 16 + r) * 136 + kc * 32 + 16 + hh * 8);
                aq.q[0] = pa0[0];
                aq.q[1] = pa1[0];
                const uint4* pb0 = (const uint4*)(Ks + (snt * 16 + r) * 136 + kc * 32 + hh * 8);
                const uint4* pb1 = (const uint4*)(Ks + (snt * 16 + r) * 136 + kc * 32 + 16 + hh * 8);
                bk.q[0] = pb0[0];
                bk.q[1] = pb1[0];
                sacc = __builtin_amdgcn_wmma_f32_16x16x32_bf16(
                    false, aq.v, false, bk.v, (short)0, sacc, false, false);
            }
#pragma unroll
            for (int i = 0; i < 8; ++i) {
                const int rowl = i + hh * 8;
                const int grow = qBase + omt * 16 + rowl;
                const int gcol = kBase + snt * 16 + r;
                float val = sacc[i] * rs;
                if (gcol > grow) val = -3.0e38f;
                Ss[(omt * 16 + rowl) * 72 + snt * 16 + r] = val;
            }
        }
        __syncthreads();

        // ---- online softmax: 4 threads per row, combine via shfl_xor ----
        {
            const int sr = t >> 2;      // row 0..63
            const int sg = t & 3;       // 16-col segment
            const float mo = mrow[sr];
            float mx = -3.0e38f;
#pragma unroll
            for (int c = 0; c < 16; ++c)
                mx = fmaxf(mx, Ss[sr * 72 + sg * 16 + c]);
            mx = fmaxf(mx, __shfl_xor(mx, 1, 32));
            mx = fmaxf(mx, __shfl_xor(mx, 2, 32));
            mx = fmaxf(mx, mo);
            float sm = 0.f;
#pragma unroll
            for (int c = 0; c < 16; ++c) {
                const float p = __expf(Ss[sr * 72 + sg * 16 + c] - mx);
                sm += p;
                Ps[sr * 72 + sg * 16 + c] = f2bf(p);
            }
            sm += __shfl_xor(sm, 1, 32);
            sm += __shfl_xor(sm, 2, 32);
            if (sg == 0) {
                const float al = __expf(mo - mx);
                lrow[sr] = lrow[sr] * al + sm;
                mrow[sr] = mx;
                arow[sr] = al;
            }
        }
        __syncthreads();

        // ---- rescale O fragments, then O += P @ V ----
#pragma unroll
        for (int nt = 0; nt < 4; ++nt) {
#pragma unroll
            for (int i = 0; i < 8; ++i)
                oacc[nt][i] *= arow[omt * 16 + i + hh * 8];
#pragma unroll
            for (int kc = 0; kc < 2; ++kc) {
                FragB ap, bv;
                const uint4* pa0 = (const uint4*)(Ps + (omt * 16 + r) * 72 + kc * 32 + hh * 8);
                const uint4* pa1 = (const uint4*)(Ps + (omt * 16 + r) * 72 + kc * 32 + 16 + hh * 8);
                ap.q[0] = pa0[0];
                ap.q[1] = pa1[0];
                // Vt[col][key]: keys [hh*8..+7] and [16+hh*8..+7] per lane -> 2x ds_load_b128
                const uint4* pv0 = (const uint4*)(Vt + (onb + nt * 16 + r) * 72 + kc * 32 + hh * 8);
                const uint4* pv1 = (const uint4*)(Vt + (onb + nt * 16 + r) * 72 + kc * 32 + 16 + hh * 8);
                bv.q[0] = pv0[0];
                bv.q[1] = pv1[0];
                oacc[nt] = __builtin_amdgcn_wmma_f32_16x16x32_bf16(
                    false, ap.v, false, bv.v, (short)0, oacc[nt], false, false);
            }
        }
    }
    __syncthreads();

    // ---- O / l -> global (bf16) ----
#pragma unroll
    for (int nt = 0; nt < 4; ++nt)
#pragma unroll
        for (int i = 0; i < 8; ++i) {
            const int rowl = omt * 16 + i + hh * 8;
            const float inv = 1.f / lrow[rowl];
            const int grow = qBase + rowl;
            const int gcol = onb + nt * 16 + r;
            ob[bbase + (size_t)grow * DMODEL + hoff + gcol] = f2bf(oacc[nt][i] * inv);
        }
}

// ---------------------------------------------------------------------------
// modhA[token][r] = 0.1 * tanh( (attn @ A)[r] ) * (h @ A)[r]      (R = 16)
// ---------------------------------------------------------------------------
__global__ __launch_bounds__(256) void mod_lora(
        const unsigned short* __restrict__ attn,
        const unsigned short* __restrict__ hbuf,
        const float* __restrict__ Amat,
        float* __restrict__ out) {
    const int tok = blockIdx.x;
    const int t = threadIdx.x, rr = t & 15, ch = t >> 4;
    const size_t base = (size_t)tok * DMODEL;
    float sa = 0.f, sh = 0.f;
    for (int i = 0; i < 128; ++i) {
        const int d = ch * 128 + i;
        const float a = Amat[(size_t)d * RLORA + rr];
        sa += bf2f(attn[base + d]) * a;
        sh += bf2f(hbuf[base + d]) * a;
    }
    __shared__ float ra[16][16], rh[16][16];
    ra[ch][rr] = sa;
    rh[ch][rr] = sh;
    __syncthreads();
    if (t < 16) {
        float SA = 0.f, SH = 0.f;
#pragma unroll
        for (int c = 0; c < 16; ++c) { SA += ra[c][t]; SH += rh[c][t]; }
        out[(size_t)tok * RLORA + t] = 0.1f * tanhf(SA) * SH;
    }
}

// ---------------------------------------------------------------------------
extern "C" void kernel_launch(void* const* d_in, const int* in_sizes, int n_in,
                              void* d_out, int out_size, void* d_ws, size_t ws_size,
                              hipStream_t stream) {
    (void)in_sizes; (void)n_in; (void)out_size; (void)ws_size;
    const float* hidden = (const float*)d_in[0];
    const float* ln1    = (const float*)d_in[1];
    const float* ln2    = (const float*)d_in[2];
    const float* Wq     = (const float*)d_in[3];
    const float* Wk     = (const float*)d_in[4];
    const float* Wv     = (const float*)d_in[5];
    const float* Wo     = (const float*)d_in[6];
    const float* Amat   = (const float*)d_in[7];
    const float* Bmat   = (const float*)d_in[8];
    const float* Win    = (const float*)d_in[9];
    const float* Wup    = (const float*)d_in[10];
    const float* Wdown  = (const float*)d_in[11];

    const int M = BATCH * SEQ;                                  // 4096
    char* ws = (char*)d_ws;
    const size_t TOKD = (size_t)M * DMODEL * sizeof(unsigned short);   // 16.8 MB
    const size_t TOKF = (size_t)M * FDIM * sizeof(unsigned short);     // 67 MB

    unsigned short* xln   = (unsigned short*)(ws);              // ln1(x) bf16
    unsigned short* qbuf  = (unsigned short*)(ws + 1 * TOKD);
    unsigned short* kbuf  = (unsigned short*)(ws + 2 * TOKD);
    unsigned short* vbuf  = (unsigned short*)(ws + 3 * TOKD);
    unsigned short* oh    = (unsigned short*)(ws + 4 * TOKD);   // attn heads out
    unsigned short* attnb = xln;                                // reuse (post-Wo)
    unsigned short* hb    = qbuf;                               // reuse (ln2)
    float*          modhA = (float*)(ws + 2 * TOKD);            // reuse kbuf slot
    unsigned short* upb   = (unsigned short*)(ws + 5 * TOKD);
    unsigned short* actb  = (unsigned short*)(ws + 5 * TOKD + TOKF);

    const dim3 blk(256);

    // 1) ln1
    rmsnorm_bf16<<<M, blk, 0, stream>>>(hidden, ln1, xln);

    // 2) Q/K/V projections
    const dim3 gdd(DMODEL / 128, M / 128);
    gemm_bf16<EPI_BF16><<<gdd, blk, 0, stream>>>(xln, Wq, qbuf, M, DMODEL, DMODEL,
                                                 nullptr, nullptr, nullptr, nullptr);
    gemm_bf16<EPI_BF16><<<gdd, blk, 0, stream>>>(xln, Wk, kbuf, M, DMODEL, DMODEL,
                                                 nullptr, nullptr, nullptr, nullptr);
    gemm_bf16<EPI_BF16><<<gdd, blk, 0, stream>>>(xln, Wv, vbuf, M, DMODEL, DMODEL,
                                                 nullptr, nullptr, nullptr, nullptr);

    // 3) flash attention
    const dim3 gattn(SEQ / 64, NHEAD, BATCH);
    const size_t smem = (size_t)(2 * 64 * 136 * 2 +     // Qs, Ks
                                 128 * 72 * 2 +         // Vt
                                 64 * 72 * 2 +          // Ps
                                 64 * 72 * 4 +          // Ss
                                 3 * 64 * 4);           // mrow/lrow/arow
    flash_attn<<<gattn, blk, smem, stream>>>(qbuf, kbuf, vbuf, oh);

    // 4) output projection
    gemm_bf16<EPI_BF16><<<gdd, blk, 0, stream>>>(oh, Wo, attnb, M, DMODEL, DMODEL,
                                                 nullptr, nullptr, nullptr, nullptr);

    // 5) ln2 on ORIGINAL residual (NPT quirk)
    rmsnorm_bf16<<<M, blk, 0, stream>>>(hidden, ln2, hb);

    // 6) modhA = 0.1*tanh(attn@A) * (h@A)
    mod_lora<<<M, blk, 0, stream>>>(attnb, hb, Amat, modhA);

    // 7) up projection
    const dim3 gdf(FDIM / 128, M / 128);
    gemm_bf16<EPI_BF16><<<gdf, blk, 0, stream>>>(hb, Wup, upb, M, FDIM, DMODEL,
                                                 nullptr, nullptr, nullptr, nullptr);

    // 8) gate = silu(h@Win^T + (modhA)@B) * up   -> act bf16
    gemm_bf16<EPI_GATE><<<gdf, blk, 0, stream>>>(hb, Win, actb, M, FDIM, DMODEL,
                                                 modhA, Bmat, upb, nullptr);

    // 9) out = residual + act @ Wdown^T          -> f32
    const dim3 gfd(DMODEL / 128, M / 128);
    gemm_bf16<EPI_RESID><<<gfd, blk, 0, stream>>>(actb, Wdown, d_out, M, DMODEL, FDIM,
                                                  nullptr, nullptr, nullptr, hidden);
}
